// DTSP_32633161515436
// MI455X (gfx1250) — matrix operand order repaired and checked by reading, not compiled, not run on hardware
//
#include <hip/hip_runtime.h>
#include <hip/hip_bf16.h>

// ---------------------------------------------------------------------------
// Types for CDNA5 WMMA (wave32, 16x16x32 f16 -> f32)
// ---------------------------------------------------------------------------
typedef _Float16 f16_t;
typedef __attribute__((ext_vector_type(16))) _Float16 v16h;
typedef __attribute__((ext_vector_type(8)))  _Float16 v8h;
typedef __attribute__((ext_vector_type(8)))  float    v8f;

#define DIM     64
#define NUM_MP  32
#define B_INST  8
#define V_PER   100
#define E_PER   4950
#define E_TOTAL 39600
#define V_TOTAL 800
#define E_TILES (E_TOTAL / 16)   // 2475
#define V_TILES (V_TOTAL / 16)   // 50

__device__ __forceinline__ void ds_fence() {
  // wave-local ordering of LDS store->load round trips (cross-lane via LDS):
  // hardware keeps same-wave DS ops in order; the wait + memory clobber stops
  // the compiler from reordering around it.
  asm volatile("s_wait_dscnt 0x0" ::: "memory");
}

__device__ __forceinline__ v8f wmma_f16(v16h a, v16h b, v8f c) {
  return __builtin_amdgcn_wmma_f32_16x16x32_f16(
      /*neg_a=*/false, a, /*neg_b=*/false, b,
      /*c_mod=*/(short)0, c, /*reuse_a=*/false, /*reuse_b=*/false);
}

// Load a 16x32 A tile from row-major f16 (global or LDS), rows row0..row0+15,
// K columns k0..k0+31.  ISA A layout: lane L holds row L&15; element e has
// K = k0 + e + 8*(L>>4) + (e>=8 ? 8 : 0)  => two contiguous b128 chunks.
__device__ __forceinline__ v16h load_A(const f16_t* base, int stride, int k0) {
  const int lane = threadIdx.x & 31;
  const int r = lane & 15, hh = lane >> 4;
  const f16_t* p = base + r * stride + k0 + 8 * hh;
  v8h lo = *(const v8h*)p;
  v8h hi = *(const v8h*)(p + 16);
  v16h a;
#pragma unroll
  for (int i = 0; i < 8; ++i) { a[i] = lo[i]; a[i + 8] = hi[i]; }
  return a;
}

// Load a 32x16 B tile from the pre-packed weight buffer.
// Packed layout: tile (kt,nt) -> 32 lanes x 16 f16 contiguous per lane.
// ISA B layout assumed: lane L holds column N = L&15; element e has
// K = (e&7) + 8*(L>>4) + 16*(e>>3).
__device__ __forceinline__ v16h load_B(const f16_t* packed, int ntiles, int kt, int nt) {
  const int lane = threadIdx.x & 31;
  const f16_t* p = packed + ((((size_t)(kt * ntiles + nt)) * 32 + lane) << 4);
  v8h lo = *(const v8h*)p;
  v8h hi = *(const v8h*)(p + 8);
  v16h b;
#pragma unroll
  for (int i = 0; i < 8; ++i) { b[i] = lo[i]; b[i + 8] = hi[i]; }
  return b;
}

__device__ __forceinline__ float fast_sigmoid(float x) {
  return 1.f / (1.f + __expf(-x));
}
__device__ __forceinline__ float fast_tanh(float x) {
  float e = __expf(-2.f * fabsf(x));
  float r = (1.f - e) / (1.f + e);
  return x < 0.f ? -r : r;
}

// One 16-row x 64-col dense layer: out = [relu](A @ W + b), W packed (K=64,N=64).
// D layout: element i -> row i+8*(lane>>4), col (lane&15)+16*nt.
__device__ __forceinline__ void mlp_layer(v16h A0, v16h A1,
                                          const f16_t* Wp, const float* bias,
                                          bool do_relu, f16_t* out, int out_stride) {
  const int lane = threadIdx.x & 31;
  const int n16 = lane & 15, hh = lane >> 4;
#pragma unroll
  for (int nt = 0; nt < 4; ++nt) {
    float bv = bias[nt * 16 + n16];
    v8f c;
#pragma unroll
    for (int i = 0; i < 8; ++i) c[i] = bv;
    c = wmma_f16(A0, load_B(Wp, 4, 0, nt), c);
    c = wmma_f16(A1, load_B(Wp, 4, 1, nt), c);
#pragma unroll
    for (int i = 0; i < 8; ++i) {
      float v = c[i];
      if (do_relu) v = v > 0.f ? v : 0.f;
      out[(i + 8 * hh) * out_stride + nt * 16 + n16] = (f16_t)v;
    }
  }
}

// One LSTM cell step for a 16-row tile.  x in LDS (16x64 f16 row-major),
// h state in global f16 row-major, gates = x@Wih + h@Whh + (bih+bhh),
// i/f/g/o are column tiles {d, d+4, d+8, d+12} processed per dim-chunk d.
__device__ __forceinline__ void lstm_tile(const f16_t* xlds,
                                          const f16_t* hstate, int row0,
                                          const f16_t* wihp, const f16_t* whhp,
                                          const float* bih, const float* bhh,
                                          float* cstate, f16_t* hout, f16_t* hlds) {
  const int lane = threadIdx.x & 31;
  const int n16 = lane & 15, hh = lane >> 4;
  v16h Ax0 = load_A(xlds, DIM, 0);
  v16h Ax1 = load_A(xlds, DIM, 32);
  v16h Ah0 = load_A(hstate + (size_t)row0 * DIM, DIM, 0);
  v16h Ah1 = load_A(hstate + (size_t)row0 * DIM, DIM, 32);
#pragma unroll
  for (int d = 0; d < 4; ++d) {
    v8f g[4];
#pragma unroll
    for (int q = 0; q < 4; ++q) {
      int nt = q * 4 + d;
      float bv = bih[nt * 16 + n16] + bhh[nt * 16 + n16];
      v8f c;
#pragma unroll
      for (int i = 0; i < 8; ++i) c[i] = bv;
      c = wmma_f16(Ax0, load_B(wihp, 16, 0, nt), c);
      c = wmma_f16(Ax1, load_B(wihp, 16, 1, nt), c);
      c = wmma_f16(Ah0, load_B(whhp, 16, 0, nt), c);
      c = wmma_f16(Ah1, load_B(whhp, 16, 1, nt), c);
      g[q] = c;
    }
#pragma unroll
    for (int i = 0; i < 8; ++i) {
      int row = row0 + i + 8 * hh;
      int dim = d * 16 + n16;
      size_t idx = (size_t)row * DIM + dim;
      float iv = fast_sigmoid(g[0][i]);
      float fv = fast_sigmoid(g[1][i]);
      float gv = fast_tanh(g[2][i]);
      float ov = fast_sigmoid(g[3][i]);
      float cn = fv * cstate[idx] + iv * gv;
      cstate[idx] = cn;
      float hn = ov * fast_tanh(cn);
      hout[idx] = (f16_t)hn;
      if (hlds) hlds[(i + 8 * hh) * DIM + dim] = (f16_t)hn;
    }
  }
}

// ---------------------------------------------------------------------------
// Setup kernels
// ---------------------------------------------------------------------------

// Pack f32 weight W[K][N] (row-major) into WMMA B-tile layout (f16).
__global__ void k_pack(const float* __restrict__ src, f16_t* __restrict__ dst,
                       int K, int N) {
  int t = blockIdx.x * blockDim.x + threadIdx.x;
  if (t >= K * N) return;
  int e = t & 15;
  int lane = (t >> 4) & 31;
  int tileid = t >> 9;
  int ntiles = N >> 4;
  int nt = tileid % ntiles;
  int kt = tileid / ntiles;
  int k = kt * 32 + (e & 7) + 8 * (lane >> 4) + ((e >> 3) << 4);
  int n = nt * 16 + (lane & 15);
  dst[t] = (f16_t)src[k * N + n];
}

// Edge -> (i,j) endpoints for the complete graph on V_PER vertices.
__global__ void k_ep(int2* __restrict__ ep) {
  int e = blockIdx.x * blockDim.x + threadIdx.x;
  if (e >= E_PER) return;
  int i = 0, rem = e;
  while (rem >= (V_PER - 1) - i) { rem -= (V_PER - 1) - i; ++i; }
  ep[e] = make_int2(i, i + 1 + rem);
}

// Vertex -> its 99 incident edge ids (local).
__global__ void k_inc(int* __restrict__ inc) {
  int v = blockIdx.x * blockDim.x + threadIdx.x;
  if (v >= V_PER) return;
  int* w = inc + v * (V_PER - 1);
  int c = 0;
  for (int i = 0; i < v; ++i)
    w[c++] = i * V_PER - i * (i + 1) / 2 + (v - i - 1);
  for (int j = v + 1; j < V_PER; ++j)
    w[c++] = v * V_PER - v * (v + 1) / 2 + (j - v - 1);
}

// Edge embedding init: 4-layer MLP 2->8->16->32->64 on [W, C], scalar VALU.
__global__ void k_initE(const float* __restrict__ Wf, const float* __restrict__ Cc,
                        const float* iw0, const float* ib0,
                        const float* iw1, const float* ib1,
                        const float* iw2, const float* ib2,
                        const float* iw3, const float* ib3,
                        f16_t* __restrict__ E, float* __restrict__ cE) {
  int e = blockIdx.x * blockDim.x + threadIdx.x;
  if (e >= E_TOTAL) return;
  float x0 = Wf[e], x1 = Cc[e];
  float h1[8], h2[16], h3[32];
#pragma unroll
  for (int n = 0; n < 8; ++n) {
    float s = x0 * iw0[n] + x1 * iw0[8 + n] + ib0[n];
    h1[n] = s > 0.f ? s : 0.f;
  }
#pragma unroll
  for (int n = 0; n < 16; ++n) {
    float s = ib1[n];
#pragma unroll
    for (int k = 0; k < 8; ++k) s += h1[k] * iw1[k * 16 + n];
    h2[n] = s > 0.f ? s : 0.f;
  }
#pragma unroll
  for (int n = 0; n < 32; ++n) {
    float s = ib2[n];
#pragma unroll
    for (int k = 0; k < 16; ++k) s += h2[k] * iw2[k * 32 + n];
    h3[n] = s > 0.f ? s : 0.f;
  }
  for (int n = 0; n < DIM; ++n) {
    float s = ib3[n];
#pragma unroll
    for (int k = 0; k < 32; ++k) s += h3[k] * iw3[k * DIM + n];
    E[(size_t)e * DIM + n] = (f16_t)s;
    cE[(size_t)e * DIM + n] = 0.f;
  }
}

__global__ void k_initV(const float* __restrict__ vinit,
                        f16_t* __restrict__ V, float* __restrict__ cV) {
  int t = blockIdx.x * blockDim.x + threadIdx.x;
  if (t >= V_TOTAL * DIM) return;
  V[t] = (f16_t)(vinit[t & (DIM - 1)] * 0.125f);  // / sqrt(64)
  cV[t] = 0.f;
}

// ---------------------------------------------------------------------------
// Message-passing kernels (one 16-row tile per wave, 8 waves per block)
// ---------------------------------------------------------------------------

// msgE = mlp3_em(E)
__global__ void k_edge_mlp(const f16_t* __restrict__ E,
                           const f16_t* em0p, const f16_t* em1p, const f16_t* em2p,
                           const float* b0, const float* b1, const float* b2,
                           f16_t* __restrict__ msgE) {
  __shared__ f16_t lsh[8][16 * DIM];
  int wave = threadIdx.x >> 5;
  int tile = blockIdx.x * 8 + wave;
  if (tile >= E_TILES) return;  // wave-uniform: EXEC all-1 inside
  f16_t* my = lsh[wave];
  size_t row0 = (size_t)tile * 16;
  v16h A0 = load_A(E + row0 * DIM, DIM, 0);
  v16h A1 = load_A(E + row0 * DIM, DIM, 32);
  mlp_layer(A0, A1, em0p, b0, true, my, DIM);
  ds_fence();
  A0 = load_A(my, DIM, 0); A1 = load_A(my, DIM, 32);
  ds_fence();
  mlp_layer(A0, A1, em1p, b1, true, my, DIM);
  ds_fence();
  A0 = load_A(my, DIM, 0); A1 = load_A(my, DIM, 32);
  ds_fence();
  mlp_layer(A0, A1, em2p, b2, false, msgE + row0 * DIM, DIM);
}

// xV = gather(msgE); V,cV = LSTM_v(xV); msgV = mlp3_vm(V)
__global__ void k_vertex(const f16_t* __restrict__ msgE,
                         const int* __restrict__ inc,
                         f16_t* V, float* cV,
                         const f16_t* wihp, const f16_t* whhp,
                         const float* bih, const float* bhh,
                         const f16_t* vm0p, const f16_t* vm1p, const f16_t* vm2p,
                         const float* vb0, const float* vb1, const float* vb2,
                         f16_t* __restrict__ msgV) {
  __shared__ f16_t xsh[8][16 * DIM];
  __shared__ f16_t hsh[8][16 * DIM];
  int wave = threadIdx.x >> 5;
  int tile = blockIdx.x * 8 + wave;
  if (tile >= V_TILES) return;
  int lane = threadIdx.x & 31;
  int r = lane & 15, hh = lane >> 4;
  int v = tile * 16 + r;
  int b = v / V_PER, vl = v % V_PER;
  int d0 = hh * 32;  // each lane accumulates 32 of the 64 dims of its row
  float acc[32];
#pragma unroll
  for (int j = 0; j < 32; ++j) acc[j] = 0.f;
  const int* myinc = inc + vl * (V_PER - 1);
  for (int k = 0; k < V_PER - 1; ++k) {
    size_t ge = (size_t)b * E_PER + myinc[k];
    const v8h* p = (const v8h*)(msgE + ge * DIM + d0);
#pragma unroll
    for (int c4 = 0; c4 < 4; ++c4) {
      v8h q = p[c4];
#pragma unroll
      for (int j = 0; j < 8; ++j) acc[c4 * 8 + j] += (float)q[j];
    }
  }
  f16_t* myx = xsh[wave];
#pragma unroll
  for (int j = 0; j < 32; ++j) myx[r * DIM + d0 + j] = (f16_t)acc[j];
  ds_fence();
  lstm_tile(myx, V, tile * 16, wihp, whhp, bih, bhh, cV, V, hsh[wave]);
  ds_fence();
  v16h A0 = load_A(hsh[wave], DIM, 0);
  v16h A1 = load_A(hsh[wave], DIM, 32);
  ds_fence();
  mlp_layer(A0, A1, vm0p, vb0, true, myx, DIM);
  ds_fence();
  A0 = load_A(myx, DIM, 0); A1 = load_A(myx, DIM, 32);
  ds_fence();
  mlp_layer(A0, A1, vm1p, vb1, true, myx, DIM);
  ds_fence();
  A0 = load_A(myx, DIM, 0); A1 = load_A(myx, DIM, 32);
  ds_fence();
  mlp_layer(A0, A1, vm2p, vb2, false, msgV + (size_t)tile * 16 * DIM, DIM);
}

// xE = msgV[i] + msgV[j]; E,cE = LSTM_e(xE)
__global__ void k_edge_lstm(const f16_t* __restrict__ msgV,
                            const int2* __restrict__ ep,
                            f16_t* E, float* cE,
                            const f16_t* wihp, const f16_t* whhp,
                            const float* bih, const float* bhh) {
  __shared__ f16_t xsh[8][16 * DIM];
  int wave = threadIdx.x >> 5;
  int tile = blockIdx.x * 8 + wave;
  if (tile >= E_TILES) return;
  int lane = threadIdx.x & 31;
  int r = lane & 15, hh = lane >> 4;
  int e = tile * 16 + r;
  int b = e / E_PER, el = e % E_PER;
  int2 pv = ep[el];
  size_t vi = (size_t)(b * V_PER + pv.x);
  size_t vj = (size_t)(b * V_PER + pv.y);
  int d0 = hh * 32;
  const v8h* pi = (const v8h*)(msgV + vi * DIM + d0);
  const v8h* pj = (const v8h*)(msgV + vj * DIM + d0);
  f16_t* myx = xsh[wave];
#pragma unroll
  for (int c4 = 0; c4 < 4; ++c4) {
    v8h qa = pi[c4], qb = pj[c4];
#pragma unroll
    for (int j = 0; j < 8; ++j)
      myx[r * DIM + d0 + c4 * 8 + j] = (f16_t)((float)qa[j] + (float)qb[j]);
  }
  ds_fence();
  lstm_tile(myx, E, tile * 16, wihp, whhp, bih, bhh, cE, E, (f16_t*)nullptr);
}

// votes = mlp3_vt(E)[:,0]
__global__ void k_vote(const f16_t* __restrict__ E,
                       const f16_t* vt0p, const f16_t* vt1p,
                       const float* b0, const float* b1,
                       const float* w2, const float* b2,
                       float* __restrict__ votes) {
  __shared__ f16_t lsh[8][16 * DIM];
  int wave = threadIdx.x >> 5;
  int tile = blockIdx.x * 8 + wave;
  if (tile >= E_TILES) return;
  f16_t* my = lsh[wave];
  size_t row0 = (size_t)tile * 16;
  v16h A0 = load_A(E + row0 * DIM, DIM, 0);
  v16h A1 = load_A(E + row0 * DIM, DIM, 32);
  mlp_layer(A0, A1, vt0p, b0, true, my, DIM);
  ds_fence();
  A0 = load_A(my, DIM, 0); A1 = load_A(my, DIM, 32);
  ds_fence();
  mlp_layer(A0, A1, vt1p, b1, true, my, DIM);
  ds_fence();
  int lane = threadIdx.x & 31;
  if (lane < 16) {   // divergence only after all WMMA done
    float s = b2[0];
    for (int k = 0; k < DIM; ++k) s += (float)my[lane * DIM + k] * w2[k];
    votes[row0 + lane] = s;
  }
}

// Per-instance mean over its 4950 edges (deterministic block reduction).
__global__ void k_reduce(const float* __restrict__ votes, float* __restrict__ out) {
  __shared__ float s[256];
  int b = blockIdx.x;
  float a = 0.f;
  for (int i = threadIdx.x; i < E_PER; i += 256) a += votes[b * E_PER + i];
  s[threadIdx.x] = a;
  __syncthreads();
  for (int w = 128; w > 0; w >>= 1) {
    if (threadIdx.x < w) s[threadIdx.x] += s[threadIdx.x + w];
    __syncthreads();
  }
  if (threadIdx.x == 0) out[b] = s[0] / (float)E_PER;
}

// ---------------------------------------------------------------------------
// Host launch
// ---------------------------------------------------------------------------
extern "C" void kernel_launch(void* const* d_in, const int* in_sizes, int n_in,
                              void* d_out, int out_size, void* d_ws, size_t ws_size,
                              hipStream_t stream) {
  (void)in_sizes; (void)n_in; (void)out_size; (void)ws_size;
  const float* Wfeat = (const float*)d_in[1];
  const float* Cfeat = (const float*)d_in[2];
  const float* iw0 = (const float*)d_in[4];  const float* ib0 = (const float*)d_in[5];
  const float* iw1 = (const float*)d_in[6];  const float* ib1 = (const float*)d_in[7];
  const float* iw2 = (const float*)d_in[8];  const float* ib2 = (const float*)d_in[9];
  const float* iw3 = (const float*)d_in[10]; const float* ib3 = (const float*)d_in[11];
  const float* em_w0 = (const float*)d_in[12]; const float* em_b0 = (const float*)d_in[13];
  const float* em_w1 = (const float*)d_in[14]; const float* em_b1 = (const float*)d_in[15];
  const float* em_w2 = (const float*)d_in[16]; const float* em_b2 = (const float*)d_in[17];
  const float* vm_w0 = (const float*)d_in[18]; const float* vm_b0 = (const float*)d_in[19];
  const float* vm_w1 = (const float*)d_in[20]; const float* vm_b1 = (const float*)d_in[21];
  const float* vm_w2 = (const float*)d_in[22]; const float* vm_b2 = (const float*)d_in[23];
  const float* vt_w0 = (const float*)d_in[24]; const float* vt_b0 = (const float*)d_in[25];
  const float* vt_w1 = (const float*)d_in[26]; const float* vt_b1 = (const float*)d_in[27];
  const float* vt_w2 = (const float*)d_in[28]; const float* vt_b2 = (const float*)d_in[29];
  const float* v_init = (const float*)d_in[30];
  const float* wih_v = (const float*)d_in[31]; const float* whh_v = (const float*)d_in[32];
  const float* bih_v = (const float*)d_in[33]; const float* bhh_v = (const float*)d_in[34];
  const float* wih_e = (const float*)d_in[35]; const float* whh_e = (const float*)d_in[36];
  const float* bih_e = (const float*)d_in[37]; const float* bhh_e = (const float*)d_in[38];

  char* base = (char*)d_ws;
  size_t off = 0;
  auto take = [&](size_t bytes) -> void* {
    void* p = base + off;
    off = (off + bytes + 255) & ~(size_t)255;
    return p;
  };
  f16_t* Est  = (f16_t*)take((size_t)E_TOTAL * DIM * sizeof(f16_t));
  float* cE   = (float*)take((size_t)E_TOTAL * DIM * sizeof(float));
  f16_t* Vst  = (f16_t*)take((size_t)V_TOTAL * DIM * sizeof(f16_t));
  float* cV   = (float*)take((size_t)V_TOTAL * DIM * sizeof(float));
  f16_t* msgE = (f16_t*)take((size_t)E_TOTAL * DIM * sizeof(f16_t));
  f16_t* msgV = (f16_t*)take((size_t)V_TOTAL * DIM * sizeof(f16_t));
  float* votes = (float*)take((size_t)E_TOTAL * sizeof(float));
  int2* ep = (int2*)take((size_t)E_PER * sizeof(int2));
  int* inc = (int*)take((size_t)V_PER * (V_PER - 1) * sizeof(int));
  f16_t* pEm0 = (f16_t*)take(64 * 64 * sizeof(f16_t));
  f16_t* pEm1 = (f16_t*)take(64 * 64 * sizeof(f16_t));
  f16_t* pEm2 = (f16_t*)take(64 * 64 * sizeof(f16_t));
  f16_t* pVm0 = (f16_t*)take(64 * 64 * sizeof(f16_t));
  f16_t* pVm1 = (f16_t*)take(64 * 64 * sizeof(f16_t));
  f16_t* pVm2 = (f16_t*)take(64 * 64 * sizeof(f16_t));
  f16_t* pVt0 = (f16_t*)take(64 * 64 * sizeof(f16_t));
  f16_t* pVt1 = (f16_t*)take(64 * 64 * sizeof(f16_t));
  f16_t* pWihV = (f16_t*)take(64 * 256 * sizeof(f16_t));
  f16_t* pWhhV = (f16_t*)take(64 * 256 * sizeof(f16_t));
  f16_t* pWihE = (f16_t*)take(64 * 256 * sizeof(f16_t));
  f16_t* pWhhE = (f16_t*)take(64 * 256 * sizeof(f16_t));

  auto packW = [&](const float* src, f16_t* dst, int K, int N) {
    int total = K * N;
    k_pack<<<(total + 255) / 256, 256, 0, stream>>>(src, dst, K, N);
  };
  packW(em_w0, pEm0, 64, 64);  packW(em_w1, pEm1, 64, 64);  packW(em_w2, pEm2, 64, 64);
  packW(vm_w0, pVm0, 64, 64);  packW(vm_w1, pVm1, 64, 64);  packW(vm_w2, pVm2, 64, 64);
  packW(vt_w0, pVt0, 64, 64);  packW(vt_w1, pVt1, 64, 64);
  packW(wih_v, pWihV, 64, 256); packW(whh_v, pWhhV, 64, 256);
  packW(wih_e, pWihE, 64, 256); packW(whh_e, pWhhE, 64, 256);

  k_ep<<<(E_PER + 255) / 256, 256, 0, stream>>>(ep);
  k_inc<<<1, 128, 0, stream>>>(inc);
  k_initE<<<(E_TOTAL + 255) / 256, 256, 0, stream>>>(
      Wfeat, Cfeat, iw0, ib0, iw1, ib1, iw2, ib2, iw3, ib3, Est, cE);
  k_initV<<<(V_TOTAL * DIM + 255) / 256, 256, 0, stream>>>(v_init, Vst, cV);

  const int egrid = (E_TILES + 7) / 8;  // 310
  const int vgrid = (V_TILES + 7) / 8;  // 7
  for (int it = 0; it < NUM_MP; ++it) {
    k_edge_mlp<<<egrid, 256, 0, stream>>>(Est, pEm0, pEm1, pEm2,
                                          em_b0, em_b1, em_b2, msgE);
    k_vertex<<<vgrid, 256, 0, stream>>>(msgE, inc, Vst, cV,
                                        pWihV, pWhhV, bih_v, bhh_v,
                                        pVm0, pVm1, pVm2, vm_b0, vm_b1, vm_b2,
                                        msgV);
    k_edge_lstm<<<egrid, 256, 0, stream>>>(msgV, ep, Est, cE,
                                           pWihE, pWhhE, bih_e, bhh_e);
  }

  k_vote<<<egrid, 256, 0, stream>>>(Est, pVt0, pVt1, vt_b0, vt_b1,
                                    vt_w2, vt_b2, votes);
  k_reduce<<<B_INST, 256, 0, stream>>>(votes, (float*)d_out);
}
